// SparseDiffAttn_55705725829376
// MI455X (gfx1250) — matrix operand order repaired and checked by reading, not compiled, not run on hardware
//
#include <hip/hip_runtime.h>
#include <hip/hip_bf16.h>
#include <stdint.h>

typedef __bf16 bf16;
typedef __attribute__((ext_vector_type(16))) bf16  v16bf;
typedef __attribute__((ext_vector_type(8)))  bf16  v8bf;
typedef __attribute__((ext_vector_type(8)))  float v8f;

constexpr int   H        = 24;
constexpr int   N        = 3072;
constexpr int   D        = 128;
constexpr int   QG       = 16;          // n / BM
constexpr int   BMQ      = 192;         // query-group block
constexpr int   TOPK_SEL = 896;
constexpr int   HND      = H * N * D;   // 9,437,184 elements per plane
constexpr int   CSN      = H * QG * N;  // colsum elements
constexpr float SCALE    = 0.08838834764831845f; // 1/sqrt(128)
constexpr int   WAVES    = 12;          // 12 waves * 16 rows = 192 queries

// round-to-nearest-even f32 -> bf16 without relying on __bf16 conversions
__device__ __forceinline__ bf16 f2bf(float f) {
  unsigned u = __builtin_bit_cast(unsigned, f);
  u += 0x7fffu + ((u >> 16) & 1u);
  unsigned short s = (unsigned short)(u >> 16);
  return __builtin_bit_cast(bf16, s);
}

__device__ __forceinline__ v8f wmma_bf16(v16bf a, v16bf b, v8f c) {
  // D = A(16x32 bf16) * B(32x16 bf16) + C(16x16 f32)
  return __builtin_amdgcn_wmma_f32_16x16x32_bf16(false, a, false, b,
                                                 (short)0, c, false, false);
}

// A-fragment (16x32 bf16) from a row-major 16x128 tile: row = lane&15,
// lane-half 0 holds K {0..15}, half 1 holds K {16..31}.
__device__ __forceinline__ v16bf load_afrag(const bf16* row, int c, int half) {
  v8bf lo = *(const v8bf*)(row + c * 32 + half * 16);
  v8bf hi = *(const v8bf*)(row + c * 32 + half * 16 + 8);
  v16bf a;
#pragma unroll
  for (int i = 0; i < 8; ++i) { a[i] = lo[i]; a[8 + i] = hi[i]; }
  return a;
}

// ---------------------------------------------------------------------------
// 1) f32 -> bf16 conversion; V stored d-major (vT[h][d][n]); zero colsums
// ---------------------------------------------------------------------------
__global__ void __launch_bounds__(256)
convert_kernel(const float* __restrict__ q, const float* __restrict__ k,
               const float* __restrict__ v, bf16* __restrict__ qb,
               bf16* __restrict__ kb, bf16* __restrict__ vT,
               float* __restrict__ cs) {
  size_t i = (size_t)blockIdx.x * blockDim.x + threadIdx.x;
  if (i < (size_t)HND) {
    qb[i] = f2bf(q[i]);
    kb[i] = f2bf(k[i]);
    size_t hh  = i / (size_t)(N * D);
    size_t rem = i - hh * (size_t)(N * D);
    int r = (int)(rem / D);
    int c = (int)(rem % D);
    vT[(hh * D + c) * N + r] = f2bf(v[i]);
  }
  if (i < (size_t)CSN) cs[i] = 0.0f;
}

// ---------------------------------------------------------------------------
// 2/4) Flash-style attention, one (head, query-group) per block; 12 waves x
//      16 queries. S is computed TRANSPOSED (S^T = K * Q^T) so per-query
//      softmax statistics are lane-local (no shuffle trees in the hot loop).
//      SPARSE=0: dense output + per-group column sums.
//      SPARSE=1: masked softmax, writes out = o_dense - sp.
// ---------------------------------------------------------------------------
template <int SPARSE>
__global__ void __launch_bounds__(384)
attn_kernel(const bf16* __restrict__ qb, const bf16* __restrict__ kb,
            const bf16* __restrict__ vT, const unsigned char* __restrict__ mask,
            float* __restrict__ cs, const float* __restrict__ o_dense,
            float* __restrict__ out) {
  __shared__ bf16  lds_p [WAVES][16][32];  // P staged bf16 for A-fragments
  __shared__ float lds_pf[WAVES][16][32];  // P staged f32 for column sums
  __shared__ float lds_cs[WAVES][32];      // per-wave colsum partials

  const int blk  = blockIdx.x;             // h * 16 + qg
  const int h    = blk >> 4;
  const int qgi  = blk & 15;
  const int tid  = threadIdx.x;
  const int w    = tid >> 5;               // wave id (0..11): 16-query tile
  const int lane = tid & 31;
  const int half = lane >> 4;
  const int col  = lane & 15;              // query within tile / C column

  const bf16* kbh = kb + (size_t)h * N * D;
  const bf16* vTh = vT + (size_t)h * D * N;
  const unsigned char* mrow = SPARSE ? (mask + (size_t)blk * N) : nullptr;
  float* csrow = (!SPARSE) ? (cs + (size_t)blk * N) : nullptr;

  // --- Q^T B-fragments (B = Q^T: K = d, N = query), contiguous 32B loads ---
  const bf16* qrow = qb + ((size_t)h * N + (size_t)qgi * BMQ + w * 16 + col) * D;
  v16bf bq[4];
#pragma unroll
  for (int c = 0; c < 4; ++c)
    bq[c] = *(const v16bf*)(qrow + c * 32 + half * 16);

  // --- pass 1: online softmax stats; m/l are per-lane scalars -------------
  // S^T C-layout: lane = query (col), VGPR elem v = key (v + 8*half).
  float mq = -1e30f, lq = 0.0f;

  for (int kt = 0; kt < N / 16; ++kt) {
    v8f s = {0.f, 0.f, 0.f, 0.f, 0.f, 0.f, 0.f, 0.f};
#pragma unroll
    for (int c = 0; c < 4; ++c) {
      const bf16* arow = kbh + (size_t)(kt * 16 + col) * D; // A row = key
      s = wmma_bf16(load_afrag(arow, c, half), bq[c], s);
    }
    uint64_t m8 = ~0ull;
    if (SPARSE) m8 = *(const uint64_t*)(mrow + kt * 16 + half * 8);
    float x[8];
    float tmax = -1e30f;
#pragma unroll
    for (int v = 0; v < 8; ++v) {
      bool valid = ((m8 >> (8 * v)) & 0xffull) != 0;
      x[v] = valid ? s[v] * SCALE : -1e30f;
      tmax = fmaxf(tmax, x[v]);
    }
    tmax = fmaxf(tmax, __shfl_xor(tmax, 16, 32));  // merge key halves
    float nm = fmaxf(mq, tmax);
    float te = 0.0f;
#pragma unroll
    for (int v = 0; v < 8; ++v) te += __expf(x[v] - nm);
    te += __shfl_xor(te, 16, 32);
    lq = lq * __expf(mq - nm) + te;
    mq = nm;
  }
  const float rlq = (lq > 0.0f) ? (1.0f / lq) : 0.0f;

  // --- pass 2: P = exp(s-m)/l, O += P*V, column sums -----------------------
  const v8f vzero = {0.f, 0.f, 0.f, 0.f, 0.f, 0.f, 0.f, 0.f};
  v8f oacc[8];
#pragma unroll
  for (int dt = 0; dt < 8; ++dt) oacc[dt] = vzero;

  for (int cb = 0; cb < N / 32; ++cb) {
#pragma unroll
    for (int sub = 0; sub < 2; ++sub) {
      const int kt = cb * 2 + sub;
      v8f s = vzero;
#pragma unroll
      for (int c = 0; c < 4; ++c) {
        const bf16* arow = kbh + (size_t)(kt * 16 + col) * D;
        s = wmma_bf16(load_afrag(arow, c, half), bq[c], s);
      }
      uint64_t m8 = ~0ull;
      if (SPARSE) m8 = *(const uint64_t*)(mrow + kt * 16 + half * 8);
#pragma unroll
      for (int v = 0; v < 8; ++v) {
        bool valid = ((m8 >> (8 * v)) & 0xffull) != 0;
        float xx = valid ? s[v] * SCALE : -1e30f;
        float p  = __expf(xx - mq) * rlq;
        int key  = sub * 16 + v + 8 * half;  // key within 32-chunk
        lds_p [w][col][key] = f2bf(p);       // row-major P for A-fragment
        if (!SPARSE) lds_pf[w][col][key] = p;
      }
    }
    __syncthreads();

    if (!SPARSE) {
      // per-wave colsum: lane kk sums its key column over 16 queries (f32)
      float acc = 0.0f;
#pragma unroll
      for (int qi = 0; qi < 16; ++qi) acc += lds_pf[w][qi][lane];
      lds_cs[w][lane] = acc;
    }

    // re-swizzle staged P into an A-fragment (16 queries x 32 keys)
    v16bf ap;
#pragma unroll
    for (int e = 0; e < 16; ++e) {
      int kk = half * 16 + e;
      ap[e] = lds_p[w][col][kk];
    }
#pragma unroll
    for (int dt = 0; dt < 8; ++dt) {
      // V B-fragment from d-major vT: lane = output d col, elem e -> key
      const bf16* vp = vTh + (size_t)(dt * 16 + col) * N + cb * 32 + half * 16;
      oacc[dt] = wmma_bf16(ap, *(const v16bf*)vp, oacc[dt]);
    }
    __syncthreads();

    if (!SPARSE && w == 0) {  // deterministic in-order colsum accumulation
      float acc = 0.0f;
#pragma unroll
      for (int ww = 0; ww < WAVES; ++ww) acc += lds_cs[ww][lane];
      csrow[cb * 32 + lane] += acc;
    }
  }

  // --- writeback: O C-layout row = vgpr + 8*half, col = lane&15 ------------
#pragma unroll
  for (int dt = 0; dt < 8; ++dt) {
#pragma unroll
    for (int i = 0; i < 8; ++i) {
      int qg_row = qgi * BMQ + w * 16 + i + 8 * half;
      int dcol   = dt * 16 + col;
      size_t idx = ((size_t)h * N + qg_row) * D + dcol;
      if (!SPARSE) out[idx] = oacc[dt][i];
      else         out[idx] = o_dense[idx] - oacc[dt][i];
    }
  }
}

// ---------------------------------------------------------------------------
// 3) top-k threshold by bisection + ~1% hash-random mask per (h, qg) row
// ---------------------------------------------------------------------------
__global__ void __launch_bounds__(256)
topk_mask_kernel(const float* __restrict__ cs, unsigned char* __restrict__ mask) {
  __shared__ float sv[N];
  __shared__ float redf[256];
  __shared__ int   redi[256];
  const int blk = blockIdx.x;
  const int t   = threadIdx.x;
  const float* row = cs + (size_t)blk * N;
  for (int i = t; i < N; i += 256) sv[i] = row[i];
  __syncthreads();

  float lmax = -1e30f, lmin = 1e30f;
  for (int i = t; i < N; i += 256) { lmax = fmaxf(lmax, sv[i]); lmin = fminf(lmin, sv[i]); }
  redf[t] = lmax; __syncthreads();
  for (int s = 128; s > 0; s >>= 1) { if (t < s) redf[t] = fmaxf(redf[t], redf[t + s]); __syncthreads(); }
  float hi = redf[0]; __syncthreads();
  redf[t] = lmin; __syncthreads();
  for (int s = 128; s > 0; s >>= 1) { if (t < s) redf[t] = fminf(redf[t], redf[t + s]); __syncthreads(); }
  float lo = redf[0]; __syncthreads();

  for (int it = 0; it < 24; ++it) {
    float mid = 0.5f * (lo + hi);
    int c = 0;
    for (int i = t; i < N; i += 256) c += (sv[i] >= mid) ? 1 : 0;
    redi[t] = c; __syncthreads();
    for (int s = 128; s > 0; s >>= 1) { if (t < s) redi[t] += redi[t + s]; __syncthreads(); }
    int cnt = redi[0]; __syncthreads();
    if (cnt > TOPK_SEL) lo = mid; else hi = mid;
  }
  float thr = lo;
  for (int i = t; i < N; i += 256) {
    unsigned hsh = (unsigned)(blk * N + i) * 2654435761u;
    hsh ^= hsh >> 13; hsh *= 0x5bd1e995u; hsh ^= hsh >> 15;
    bool rnd = (hsh % 100u) == 0u;
    mask[(size_t)blk * N + i] = ((sv[i] >= thr) || rnd) ? 1 : 0;
  }
}

// ---------------------------------------------------------------------------
extern "C" void kernel_launch(void* const* d_in, const int* in_sizes, int n_in,
                              void* d_out, int out_size, void* d_ws, size_t ws_size,
                              hipStream_t stream) {
  (void)in_sizes; (void)n_in; (void)out_size; (void)ws_size;
  const float* q = (const float*)d_in[0];
  const float* k = (const float*)d_in[1];
  const float* v = (const float*)d_in[2];
  float* out = (float*)d_out;

  // workspace layout: qb | kb | vT (bf16) | cs (f32) | mask (u8)  (~62.6 MB)
  bf16* qbp = (bf16*)d_ws;
  bf16* kbp = qbp + (size_t)HND;
  bf16* vTp = kbp + (size_t)HND;
  float* cs = (float*)(vTp + (size_t)HND);
  unsigned char* mask = (unsigned char*)(cs + (size_t)CSN);

  convert_kernel<<<dim3((HND + 255) / 256), dim3(256), 0, stream>>>(q, k, v, qbp, kbp, vTp, cs);
  attn_kernel<0><<<dim3(H * QG), dim3(384), 0, stream>>>(qbp, kbp, vTp, nullptr, cs, nullptr, out);
  topk_mask_kernel<<<dim3(H * QG), dim3(256), 0, stream>>>(cs, mask);
  attn_kernel<1><<<dim3(H * QG), dim3(384), 0, stream>>>(qbp, kbp, vTp, mask, nullptr, out, out + (size_t)HND);
}
